// VirtualNode_ConGNN_ECL_68547678045009
// MI455X (gfx1250) — matrix operand order repaired and checked by reading, not compiled, not run on hardware
//
#include <hip/hip_runtime.h>
#include <hip/hip_bf16.h>

// ---------------------------------------------------------------------------
// Problem constants (fixed by the reference's setup_inputs)
// ---------------------------------------------------------------------------
#define N_FACE 20000
#define E_FACE 320000
#define N_CTX  40000
#define E_CTX  640000
#define NG     256       // num_graphs
#define GATH   256       // GAT hidden
#define HEADS  4
#define HD     64        // head dim
#define WD     512       // whole hidden
#define VN     4
#define NC     3

typedef _Float16 f16;
typedef __attribute__((ext_vector_type(16))) _Float16 v16h;
typedef __attribute__((ext_vector_type(8)))  _Float16 v8h;
typedef __attribute__((ext_vector_type(8)))  float    v8f;

#define CDIV(a,b) (((a)+(b)-1)/(b))

// ---------------------------------------------------------------------------
// Tensor Data Mover support (CDNA5): async DMA of the A panel into LDS.
// This toolchain exposes the 6-arg builtin:
//   (uint32x4 g0, int32x8 g1, int32x4 g2, int32x4 g3, int32x8 extra, i32 cpol)
// ---------------------------------------------------------------------------
#if defined(__has_builtin)
#if __has_builtin(__builtin_amdgcn_tensor_load_to_lds)
#define HAVE_TDM 1
#endif
#endif
#ifndef HAVE_TDM
#define HAVE_TDM 0
#endif

#if HAVE_TDM
typedef __attribute__((ext_vector_type(4))) unsigned int v4u;
typedef __attribute__((ext_vector_type(8))) int          v8i;
typedef __attribute__((ext_vector_type(4))) int          v4i;

// 2D f16 tile load: tile_d0 elems/row, tile_d1 rows, row stride = stride0 elems.
// pad_enable inserts 4 DWORDs (8 halves) after every 16 DWORDs (32 halves),
// i.e. LDS row pitch = 40 halves, matching the GEMM's padded LDS layout.
__device__ __forceinline__ void tdm_load_tile_f16(
    unsigned lds_off, const void* gaddr, unsigned tensor_d0, unsigned tensor_d1,
    unsigned tile_d0, unsigned tile_d1, unsigned stride0)
{
  unsigned long long ga = (unsigned long long)(size_t)gaddr;
  v4u g0;
  g0[0] = 1u;                                   // count=1, user descriptor
  g0[1] = lds_off;                              // lds_addr  [63:32]
  g0[2] = (unsigned)(ga & 0xffffffffu);         // global_addr[31:0]
  g0[3] = (unsigned)((ga >> 32) & 0x01ffffffu)  // global_addr[56:32]
        | (2u << 30);                           // type = 2 ("image")
  v8i g1;
  g1[0] = (int)((1u << 16)      // data_size = 1 (2 bytes)
              | (1u << 20)      // pad_enable
              | (3u << 22)      // pad_interval: 16 DWORDs
              | (3u << 25));    // pad_amount: 4 DWORDs
  g1[1] = (int)((tensor_d0 & 0xffffu) << 16);                       // td0 lo
  g1[2] = (int)(((tensor_d0 >> 16) & 0xffffu) | ((tensor_d1 & 0xffffu) << 16));
  g1[3] = (int)(((tensor_d1 >> 16) & 0xffffu) | ((tile_d0 & 0xffffu) << 16));
  g1[4] = (int)(tile_d1 & 0xffffu);                                 // tile_d1, tile_d2=0
  g1[5] = (int)stride0;                                             // stride0[31:0]
  g1[6] = 0;
  g1[7] = 0;
  v4i z4 = {};
  v8i z8 = {};
  __builtin_amdgcn_tensor_load_to_lds(g0, g1, z4, z4, z8, 0);
}

__device__ __forceinline__ void tdm_wait() {
#if __has_builtin(__builtin_amdgcn_s_wait_tensorcnt)
  __builtin_amdgcn_s_wait_tensorcnt(0);
#else
  asm volatile("s_wait_tensorcnt 0x0" ::: "memory");
#endif
}
#endif

// ---------------------------------------------------------------------------
// WMMA GEMM:  C[M,N] = A[M,K](f16) @ W^T + bias
// - A staged into LDS (TDM async when available, else vector loads),
//   row-major with 40-half pitch -> A fragments are 2x ds_load_b128/lane.
// - B ("Bf") pre-swizzled in global memory into WMMA fragment order:
//   per (k-chunk kc, n-tile nt, lane) a contiguous 16-half run, so B
//   fragments are 2x global_load_b128/lane streamed through L2.
// Block = 256 threads (8 waves), tile 128x64, K-chunk 32, double-buffered.
// ---------------------------------------------------------------------------
#define BM 128
#define BN 64
#define BK 32
#define AROW 40   // padded LDS row pitch in halves (80B, multiple of 16B)

__global__ __launch_bounds__(256)
void k_gemm_f16(const f16* __restrict__ A, const f16* __restrict__ Bf,
                const float* __restrict__ bias, float* __restrict__ C,
                int M, int N, int K, int NT)
{
  __shared__ __align__(16) f16 As[2][BM][AROW];
  const int tid  = threadIdx.x;
  const int wave = tid >> 5;
  const int lane = tid & 31;
  const int m0 = blockIdx.x * BM;
  const int n0 = blockIdx.y * BN;
  const int nk = K >> 5;

  v8f acc[4] = {};

  // ---- prologue: stage K-chunk 0 into buffer 0 ----
#if HAVE_TDM
  if (wave == 0) {
    int rows = M - m0; if (rows > BM) rows = BM;
    tdm_load_tile_f16((unsigned)(size_t)&As[0][0][0], A + (size_t)m0 * K,
                      (unsigned)K, (unsigned)(M - m0), 32u, (unsigned)rows,
                      (unsigned)K);
  }
#else
  #pragma unroll
  for (int i = 0; i < 2; ++i) {
    int chunk = tid + i * 256;           // 512 chunks of 8 halves
    int r = chunk >> 2, c16 = chunk & 3;
    int gm = m0 + r;
    v8h val = {};
    if (gm < M) val = *(const v8h*)(A + (size_t)gm * K + c16 * 8);
    *(v8h*)&As[0][r][c16 * 8] = val;
  }
#endif

  for (int kc = 0; kc < nk; ++kc) {
#if HAVE_TDM
    tdm_wait();                          // buffer (kc&1) landed
#endif
    __syncthreads();
    const int cur = kc & 1;

    // stage next K-chunk into the other buffer
    if (kc + 1 < nk) {
      const int k0n = (kc + 1) * BK;
      const int nb = cur ^ 1;
#if HAVE_TDM
      if (wave == 0) {
        int rows = M - m0; if (rows > BM) rows = BM;
        tdm_load_tile_f16((unsigned)(size_t)&As[nb][0][0],
                          A + (size_t)m0 * K + k0n, (unsigned)(K - k0n),
                          (unsigned)(M - m0), 32u, (unsigned)rows, (unsigned)K);
      }
#else
      #pragma unroll
      for (int i = 0; i < 2; ++i) {
        int chunk = tid + i * 256;
        int r = chunk >> 2, c16 = chunk & 3;
        int gm = m0 + r;
        v8h val = {};
        if (gm < M) val = *(const v8h*)(A + (size_t)gm * K + k0n + c16 * 8);
        *(v8h*)&As[nb][r][c16 * 8] = val;
      }
#endif
    }

    // ---- A fragment: two wide LDS loads per lane ----
    const int mrow  = wave * 16 + (lane & 15);
    const int khalf = (lane < 16) ? 0 : 16;
    v8h alo = *(const v8h*)&As[cur][mrow][khalf];
    v8h ahi = *(const v8h*)&As[cur][mrow][khalf + 8];
    v16h a = __builtin_shufflevector(alo, ahi,
        0,1,2,3,4,5,6,7,8,9,10,11,12,13,14,15);

    // ---- B fragments: streamed from global (pre-swizzled) ----
    const f16* bbase = Bf + (((size_t)kc * NT + (n0 >> 4)) * 32 + lane) * 16;
    if (kc + 1 < nk) __builtin_prefetch(bbase + (size_t)NT * 512, 0, 0);
    #pragma unroll
    for (int nt = 0; nt < 4; ++nt) {
      const f16* bp = bbase + (size_t)nt * 512;   // 32 lanes * 16 halves
      v8h blo = *(const v8h*)bp;
      v8h bhi = *(const v8h*)(bp + 8);
      v16h b = __builtin_shufflevector(blo, bhi,
          0,1,2,3,4,5,6,7,8,9,10,11,12,13,14,15);
      acc[nt] = __builtin_amdgcn_wmma_f32_16x16x32_f16(
          false, a, false, b, (short)0, acc[nt], false, false);
    }
  }

  // ---- store per ISA C/D layout ----
  #pragma unroll
  for (int nt = 0; nt < 4; ++nt) {
    int n = n0 + nt * 16 + (lane & 15);
    if (n >= N) continue;
    float bv = bias ? bias[n] : 0.0f;
    int mbase = m0 + wave * 16 + ((lane < 16) ? 0 : 8);
    #pragma unroll
    for (int r = 0; r < 8; ++r) {
      int m = mbase + r;
      if (m < M) C[(size_t)m * N + n] = acc[nt][r] + bv;
    }
  }
}

// ---------------------------------------------------------------------------
// Weight conversion: W[N,K] f32 -> WMMA B-fragment order f16.
// Layout: Bf[ ((kc*NT + nt)*32 + lane)*16 + e ] = W[n, kc*32 + kb + e]
//   with n = nt*16 + (lane&15), kb = (lane<16) ? 0 : 16.  NT = CDIV(N,64)*4.
// ---------------------------------------------------------------------------
__global__ void k_wT_frag(const float* __restrict__ W, f16* __restrict__ Bf,
                          int N, int K, int NT) {
  size_t i = (size_t)blockIdx.x * 256 + threadIdx.x;
  size_t tot = (size_t)(K >> 5) * NT * 512;
  if (i >= tot) return;
  int e = (int)(i & 15);
  int lane = (int)((i >> 4) & 31);
  size_t rest = i >> 9;
  int nt = (int)(rest % NT);
  int kc = (int)(rest / NT);
  int n = nt * 16 + (lane & 15);
  int kb = (lane < 16) ? 0 : 16;
  int k = kc * 32 + kb + e;
  Bf[i] = (n < N && k < K) ? (f16)W[(size_t)n * K + k] : (f16)0.0f;
}

// ---------------------------------------------------------------------------
// Elementwise / helper kernels
// ---------------------------------------------------------------------------
__global__ void k_cvt_f16(const float* __restrict__ x, f16* __restrict__ y, size_t n) {
  size_t i = (size_t)blockIdx.x * 256 + threadIdx.x;
  if (i < n) y[i] = (f16)x[i];
}

__global__ void k_fill(float* __restrict__ y, float v, size_t n) {
  size_t i = (size_t)blockIdx.x * 256 + threadIdx.x;
  if (i < n) y[i] = v;
}

__global__ void k_add(const float* __restrict__ a, const float* __restrict__ b,
                      float* __restrict__ y, size_t n) {
  size_t i = (size_t)blockIdx.x * 256 + threadIdx.x;
  if (i < n) y[i] = a[i] + b[i];
}

__global__ void k_copy_cols(const float* __restrict__ x, float* __restrict__ y,
                            int M, int Ds, int Dd, int off) {
  size_t i = (size_t)blockIdx.x * 256 + threadIdx.x;
  size_t tot = (size_t)M * Ds;
  if (i >= tot) return;
  int r = (int)(i / Ds), c = (int)(i % Ds);
  y[(size_t)r * Dd + off + c] = x[i];
}

// LayerNorm (+ optional act 0/1/2 = none/relu/elu, optional residual)
__global__ __launch_bounds__(256)
void k_ln(const float* __restrict__ x, const float* __restrict__ g,
          const float* __restrict__ b, float* __restrict__ y,
          const float* __restrict__ resid, int D, int act)
{
  const int row = blockIdx.x;
  const float* xr = x + (size_t)row * D;
  __shared__ float red[256];

  float s = 0.f;
  for (int c = threadIdx.x; c < D; c += 256) s += xr[c];
  red[threadIdx.x] = s; __syncthreads();
  for (int st = 128; st > 0; st >>= 1) {
    if (threadIdx.x < st) red[threadIdx.x] += red[threadIdx.x + st];
    __syncthreads();
  }
  float mean = red[0] / (float)D;
  __syncthreads();

  float v = 0.f;
  for (int c = threadIdx.x; c < D; c += 256) { float d = xr[c] - mean; v += d * d; }
  red[threadIdx.x] = v; __syncthreads();
  for (int st = 128; st > 0; st >>= 1) {
    if (threadIdx.x < st) red[threadIdx.x] += red[threadIdx.x + st];
    __syncthreads();
  }
  float inv = rsqrtf(red[0] / (float)D + 1e-5f);

  float* yr = y + (size_t)row * D;
  const float* rr = resid ? resid + (size_t)row * D : nullptr;
  for (int c = threadIdx.x; c < D; c += 256) {
    float t = (xr[c] - mean) * inv * g[c] + b[c];
    if (act == 1) t = t > 0.f ? t : 0.f;
    else if (act == 2) t = t > 0.f ? t : (expf(t) - 1.f);
    yr[c] = (rr ? rr[c] : 0.f) + t;
  }
}

// ---------------------------------------------------------------------------
// Graph kernels
// ---------------------------------------------------------------------------
__global__ void k_edges(const int* __restrict__ ei, int E, int N,
                        int* __restrict__ s, int* __restrict__ d) {
  int i = blockIdx.x * 256 + threadIdx.x;
  int tot = E + N;
  if (i >= tot) return;
  if (i < E) { s[i] = ei[i]; d[i] = ei[E + i]; }
  else       { s[i] = i - E; d[i] = i - E; }
}

__device__ inline void atomicMaxF(float* addr, float val) {
  unsigned* u = (unsigned*)addr;
  unsigned old = __float_as_uint(*addr);
  while (__uint_as_float(old) < val) {
    unsigned prev = atomicCAS(u, old, __float_as_uint(val));
    if (prev == old) break;
    old = prev;
  }
}

__global__ void k_edge_logit(const float* __restrict__ xl, const float* __restrict__ xr,
                             const int* __restrict__ src, const int* __restrict__ dst,
                             const float* __restrict__ att,
                             float* __restrict__ logit, float* __restrict__ mx, int Etot) {
  int t = blockIdx.x * 256 + threadIdx.x;
  if (t >= Etot * HEADS) return;
  int e = t >> 2, h = t & 3;
  const float* pl = xl + (size_t)src[e] * GATH + h * HD;
  const float* pr = xr + (size_t)dst[e] * GATH + h * HD;
  const float* pa = att + h * HD;
  float acc = 0.f;
  for (int c = 0; c < HD; ++c) {
    float s = pl[c] + pr[c];
    s = s > 0.f ? s : 0.2f * s;
    acc += s * pa[c];
  }
  logit[t] = acc;
  atomicMaxF(&mx[dst[e] * HEADS + h], acc);
}

__global__ void k_edge_exp(const float* __restrict__ logit, const float* __restrict__ mx,
                           const int* __restrict__ dst, float* __restrict__ ex,
                           float* __restrict__ den, int Etot) {
  int t = blockIdx.x * 256 + threadIdx.x;
  if (t >= Etot * HEADS) return;
  int e = t >> 2, h = t & 3;
  float v = expf(logit[t] - mx[dst[e] * HEADS + h]);
  ex[t] = v;
  atomicAdd(&den[dst[e] * HEADS + h], v);
}

__global__ void k_edge_scatter(const float* __restrict__ xl, const int* __restrict__ src,
                               const int* __restrict__ dst, const float* __restrict__ ex,
                               const float* __restrict__ den, float* __restrict__ out,
                               int Etot) {
  size_t t = (size_t)blockIdx.x * 256 + threadIdx.x;
  size_t tot = (size_t)Etot * GATH;
  if (t >= tot) return;
  int e = (int)(t >> 8);
  int c = (int)(t & 255);
  int h = c >> 6;
  float alpha = ex[e * HEADS + h] / den[dst[e] * HEADS + h];
  atomicAdd(&out[(size_t)dst[e] * GATH + c], xl[(size_t)src[e] * GATH + c] * alpha);
}

// ---------------------------------------------------------------------------
// Pooling / routing / VN / GRU kernels
// ---------------------------------------------------------------------------
__global__ void k_count(const int* __restrict__ batch, float* __restrict__ cnt, int M) {
  int i = blockIdx.x * 256 + threadIdx.x;
  if (i < M) atomicAdd(&cnt[batch[i]], 1.f);
}
__global__ void k_pool_sum(const float* __restrict__ x, const int* __restrict__ batch,
                           float* __restrict__ sum, int M, int D) {
  size_t t = (size_t)blockIdx.x * 256 + threadIdx.x;
  size_t tot = (size_t)M * D;
  if (t >= tot) return;
  int i = (int)(t / D), c = (int)(t % D);
  atomicAdd(&sum[(size_t)batch[i] * D + c], x[t]);
}
__global__ void k_pool_div(float* __restrict__ sum, const float* __restrict__ cnt,
                           int D, size_t tot) {
  size_t t = (size_t)blockIdx.x * 256 + threadIdx.x;
  if (t >= tot) return;
  sum[t] = sum[t] / fmaxf(cnt[t / D], 1.f);
}

__global__ void k_head(const float* __restrict__ x, const float* __restrict__ W,
                       const float* __restrict__ b, float* __restrict__ out, int D) {
  int t = blockIdx.x * 256 + threadIdx.x;
  if (t >= NG * NC) return;
  int g = t / NC, c = t % NC;
  float acc = b[c];
  const float* xr = x + (size_t)g * D;
  const float* wr = W + (size_t)c * D;
  for (int d = 0; d < D; ++d) acc += xr[d] * wr[d];
  out[t] = acc;
}

__global__ void k_top2(const float* __restrict__ logits, float* __restrict__ w, int M) {
  int i = blockIdx.x * 256 + threadIdx.x;
  if (i >= M) return;
  float l[VN];
  for (int j = 0; j < VN; ++j) l[j] = logits[i * VN + j];
  int a = 0;
  for (int j = 1; j < VN; ++j) if (l[j] > l[a]) a = j;
  int s = -1;
  for (int j = 0; j < VN; ++j) { if (j == a) continue; if (s < 0 || l[j] > l[s]) s = j; }
  for (int j = 0; j < VN; ++j) w[i * VN + j] = (j == a || j == s) ? 1.f : 0.f;
}

__global__ void k_vn_bcast(const float* __restrict__ vinit, float* __restrict__ vn, size_t tot) {
  size_t t = (size_t)blockIdx.x * 256 + threadIdx.x;
  if (t < tot) vn[t] = vinit[t % (VN * WD)];
}

__global__ void k_vn_scatter(const float* __restrict__ w, const float* __restrict__ msg,
                             const int* __restrict__ batch, float* __restrict__ vn, int M) {
  size_t t = (size_t)blockIdx.x * 256 + threadIdx.x;
  size_t tot = (size_t)M * WD;
  if (t >= tot) return;
  int i = (int)(t / WD), c = (int)(t % WD);
  int g = batch[i];
  float m = msg[t];
  for (int j = 0; j < VN; ++j) {
    float wj = w[i * VN + j];
    if (wj != 0.f) atomicAdd(&vn[((size_t)g * VN + j) * WD + c], wj * m);
  }
}

__global__ void k_attn(const float* __restrict__ qkv, float* __restrict__ o) {
  int t = blockIdx.x * 256 + threadIdx.x;
  if (t >= NG * HEADS * VN) return;
  int g = t / (HEADS * VN);
  int h = (t / VN) % HEADS;
  int qi = t % VN;
  const int hd = WD / HEADS;                 // 128
  const float scale = 0.08838834764831845f;  // 1/sqrt(128)
  const float* q = qkv + ((size_t)g * VN + qi) * (3 * WD) + h * hd;
  float sc[VN];
  float mx = -1e30f;
  for (int j = 0; j < VN; ++j) {
    const float* kk = qkv + ((size_t)g * VN + j) * (3 * WD) + WD + h * hd;
    float acc = 0.f;
    for (int d = 0; d < hd; ++d) acc += q[d] * kk[d];
    sc[j] = acc * scale;
    mx = fmaxf(mx, sc[j]);
  }
  float den = 0.f;
  for (int j = 0; j < VN; ++j) { sc[j] = expf(sc[j] - mx); den += sc[j]; }
  for (int j = 0; j < VN; ++j) sc[j] /= den;
  float* op = o + ((size_t)g * VN + qi) * WD + h * hd;
  for (int d = 0; d < hd; ++d) {
    float acc = 0.f;
    for (int j = 0; j < VN; ++j)
      acc += sc[j] * qkv[((size_t)g * VN + j) * (3 * WD) + 2 * WD + h * hd + d];
    op[d] = acc;
  }
}

__global__ void k_recv(const float* __restrict__ w, const float* __restrict__ m2,
                       const int* __restrict__ batch, float* __restrict__ recv, int M) {
  size_t t = (size_t)blockIdx.x * 256 + threadIdx.x;
  size_t tot = (size_t)M * WD;
  if (t >= tot) return;
  int i = (int)(t / WD), c = (int)(t % WD);
  int g = batch[i];
  float acc = 0.f;
  for (int j = 0; j < VN; ++j)
    acc += w[i * VN + j] * m2[((size_t)g * VN + j) * WD + c];
  recv[t] = acc;
}

__global__ void k_gru(const float* __restrict__ gi, const float* __restrict__ gh,
                      const float* __restrict__ real, float* __restrict__ upd, int M) {
  size_t t = (size_t)blockIdx.x * 256 + threadIdx.x;
  size_t tot = (size_t)M * WD;
  if (t >= tot) return;
  int i = (int)(t / WD), c = (int)(t % WD);
  size_t b = (size_t)i * (3 * WD);
  float ir = gi[b + c], iz = gi[b + WD + c], inn = gi[b + 2 * WD + c];
  float hr = gh[b + c], hz = gh[b + WD + c], hn = gh[b + 2 * WD + c];
  float r = 1.f / (1.f + expf(-(ir + hr)));
  float z = 1.f / (1.f + expf(-(iz + hz)));
  float n = tanhf(inn + r * hn);
  upd[t] = (1.f - z) * n + z * real[t];
}

__global__ void k_rbatch(const int* __restrict__ fb, const int* __restrict__ cb,
                         int* __restrict__ rb) {
  int i = blockIdx.x * 256 + threadIdx.x;
  if (i >= N_FACE + N_CTX) return;
  rb[i] = (i < N_FACE) ? fb[i] : cb[i - N_FACE];
}

// ---------------------------------------------------------------------------
// Host orchestration
// ---------------------------------------------------------------------------
extern "C" void kernel_launch(void* const* d_in, const int* in_sizes, int n_in,
                              void* d_out, int out_size, void* d_ws, size_t ws_size,
                              hipStream_t stream) {
  (void)in_sizes; (void)n_in; (void)out_size; (void)ws_size;
  #define FIN(i) ((const float*)d_in[i])
  #define IIN(i) ((const int*)d_in[i])

  const float *fin_g = FIN(0), *fin_b = FIN(1);
  struct Gat { const float *Wp,*bp,*g0,*b0, *Wl[2],*Wr[2],*att[2],*lg[2],*lb[2]; };
  Gat fg = { FIN(2), FIN(3), FIN(4), FIN(5),
             {FIN(6), FIN(11)}, {FIN(7), FIN(12)}, {FIN(8), FIN(13)},
             {FIN(9), FIN(14)}, {FIN(10), FIN(15)} };
  Gat cg = { FIN(16), FIN(17), FIN(18), FIN(19),
             {FIN(20), FIN(25)}, {FIN(21), FIN(26)}, {FIN(22), FIN(27)},
             {FIN(23), FIN(28)}, {FIN(24), FIN(29)} };
  const float *cf_W = FIN(30), *cf_b = FIN(31), *cc_W = FIN(32), *cc_b = FIN(33);
  const float *pj_W = FIN(34), *pj_b = FIN(35), *pj_g = FIN(36), *pj_bb = FIN(37);
  const float *vn_init = FIN(38);
  const float *r_W1 = FIN(39), *r_b1 = FIN(40), *r_g = FIN(41), *r_bb = FIN(42);
  const float *r_W2 = FIN(43), *r_b2 = FIN(44);
  const float *m_W = FIN(45), *m_b = FIN(46), *m_g = FIN(47), *m_bb = FIN(48);
  const float *in_W = FIN(49), *in_b = FIN(50);
  const float *out_W = FIN(51), *out_b = FIN(52);
  const float *n_g = FIN(53), *n_b = FIN(54);
  const float *v_W = FIN(55), *v_b = FIN(56), *v_g = FIN(57), *v_bb = FIN(58);
  const float *Wih = FIN(59), *Whh = FIN(60), *bih = FIN(61), *bhh = FIN(62);
  const float *f_W1 = FIN(63), *f_b1 = FIN(64), *f_g1 = FIN(65), *f_bb1 = FIN(66);
  const float *f_W2 = FIN(67), *f_b2 = FIN(68), *f_g2 = FIN(69), *f_bb2 = FIN(70);
  const float *c_W = FIN(71), *c_b = FIN(72), *c_g = FIN(73), *c_bb = FIN(74);
  const float *w_W1 = FIN(75), *w_b1 = FIN(76), *w_g = FIN(77), *w_bb = FIN(78);
  const float *w_W2 = FIN(79), *w_b2 = FIN(80);
  const float *face_x = FIN(81), *context_x = FIN(82);
  const int *face_ei = IIN(83), *face_batch = IIN(84);
  const int *ctx_ei = IIN(85), *ctx_batch = IIN(86);

  char* base = (char*)d_ws;
  size_t off = 0;
  auto alloc = [&](size_t bytes) -> void* {
    void* p = base + off;
    off += (bytes + 255) & ~(size_t)255;
    return p;
  };
  const int NR = N_FACE + N_CTX;
  float* fxn     = (float*)alloc((size_t)N_FACE * 4096 * 4);
  f16*   fxn_h   = (f16*)  alloc((size_t)N_FACE * 4096 * 2);
  f16*   ctx_h   = (f16*)  alloc((size_t)N_CTX * 2048 * 2);
  f16*   wT      = (f16*)  alloc((size_t)4096 * 1024 * 2);   // B-fragment scratch
  float* real    = (float*)alloc((size_t)NR * GATH * 4);     // [Hf ; Hc]
  f16*   hbf     = (f16*)  alloc((size_t)N_CTX * GATH * 2);
  float* xl      = (float*)alloc((size_t)N_CTX * GATH * 4);
  float* xr      = (float*)alloc((size_t)N_CTX * GATH * 4);
  float* scr     = (float*)alloc((size_t)N_CTX * GATH * 4);
  int*   esrc    = (int*)  alloc((size_t)(E_CTX + N_CTX) * 4);
  int*   edst    = (int*)  alloc((size_t)(E_CTX + N_CTX) * 4);
  float* elog    = (float*)alloc((size_t)(E_CTX + N_CTX) * HEADS * 4);
  float* eex     = (float*)alloc((size_t)(E_CTX + N_CTX) * HEADS * 4);
  float* mx      = (float*)alloc((size_t)N_CTX * HEADS * 4);
  float* den     = (float*)alloc((size_t)N_CTX * HEADS * 4);
  int*   rbatch  = (int*)  alloc((size_t)NR * 4);
  f16*   realbf  = (f16*)  alloc((size_t)NR * GATH * 2);
  float* real512 = (float*)alloc((size_t)NR * WD * 4);
  f16*   r512bf  = (f16*)  alloc((size_t)NR * WD * 2);
  float* hlog    = (float*)alloc((size_t)NR * WD * 4);
  f16*   hlogbf  = (f16*)  alloc((size_t)NR * WD * 2);
  float* logits4 = (float*)alloc((size_t)NR * VN * 4);
  float* wts     = (float*)alloc((size_t)NR * VN * 4);
  float* msg     = (float*)alloc((size_t)NR * WD * 4);
  float* vn      = (float*)alloc((size_t)NG * VN * WD * 4);
  f16*   vnbf    = (f16*)  alloc((size_t)NG * VN * WD * 2);
  float* qkv     = (float*)alloc((size_t)NG * VN * 3 * WD * 4);
  float* ovec    = (float*)alloc((size_t)NG * VN * WD * 4);
  f16*   obf     = (f16*)  alloc((size_t)NG * VN * WD * 2);
  float* t512    = (float*)alloc((size_t)NG * VN * WD * 4);
  float* m2      = (float*)alloc((size_t)NG * VN * WD * 4);
  float* recv    = (float*)alloc((size_t)NR * WD * 4);
  f16*   recvbf  = (f16*)  alloc((size_t)NR * WD * 2);
  float* gi      = (float*)alloc((size_t)NR * 3 * WD * 4);
  float* gh      = (float*)alloc((size_t)NR * 3 * WD * 4);
  float* upd     = (float*)alloc((size_t)NR * WD * 4);
  float* whole   = (float*)alloc((size_t)NG * WD * 4);
  float* poolF   = (float*)alloc((size_t)NG * GATH * 4);
  float* poolC   = (float*)alloc((size_t)NG * GATH * 4);
  float* rf      = (float*)alloc((size_t)NG * 4096 * 4);
  f16*   rfbf    = (f16*)  alloc((size_t)NG * 4096 * 2);
  float* rf1     = (float*)alloc((size_t)NG * 1024 * 4);
  f16*   rf1bf   = (f16*)  alloc((size_t)NG * 1024 * 2);
  float* rf2     = (float*)alloc((size_t)NG * WD * 4);
  float* rc      = (float*)alloc((size_t)NG * 2048 * 4);
  f16*   rcbf    = (f16*)  alloc((size_t)NG * 2048 * 2);
  float* rc2     = (float*)alloc((size_t)NG * WD * 4);
  float* comb    = (float*)alloc((size_t)NG * 3 * WD * 4);
  f16*   combbf  = (f16*)  alloc((size_t)NG * 3 * WD * 2);
  float* w1out   = (float*)alloc((size_t)NG * WD * 4);
  float* cnt     = (float*)alloc((size_t)NG * 4);

  float* out_face  = (float*)d_out;
  float* out_ctx   = out_face + NG * NC;
  float* out_whole = out_ctx + NG * NC;

  auto gemm = [&](const f16* A, const float* W, const float* bias, float* C,
                  int M, int N, int K) {
    int NT = CDIV(N, 64) * 4;                  // 16-wide B tiles (padded)
    size_t tot = (size_t)(K >> 5) * NT * 512;
    k_wT_frag<<<CDIV(tot, (size_t)256), 256, 0, stream>>>(W, wT, N, K, NT);
    dim3 g(CDIV(M, BM), CDIV(N, BN));
    k_gemm_f16<<<g, 256, 0, stream>>>(A, wT, bias, C, M, N, K, NT);
  };
  auto cvt = [&](const float* x, f16* y, size_t n) {
    k_cvt_f16<<<CDIV(n, (size_t)256), 256, 0, stream>>>(x, y, n);
  };
  auto fill = [&](float* p, float v, size_t n) {
    k_fill<<<CDIV(n, (size_t)256), 256, 0, stream>>>(p, v, n);
  };
  auto ln = [&](const float* x, const float* g, const float* b, float* y,
                const float* resid, int M, int D, int act) {
    k_ln<<<M, 256, 0, stream>>>(x, g, b, y, resid, D, act);
  };
  auto pool = [&](const float* x, const int* batch, int M, int D, float* out) {
    fill(cnt, 0.f, NG);
    fill(out, 0.f, (size_t)NG * D);
    k_count<<<CDIV(M, 256), 256, 0, stream>>>(batch, cnt, M);
    k_pool_sum<<<CDIV((size_t)M * D, (size_t)256), 256, 0, stream>>>(x, batch, out, M, D);
    k_pool_div<<<CDIV((size_t)NG * D, (size_t)256), 256, 0, stream>>>(out, cnt, D, (size_t)NG * D);
  };
  auto run_gat = [&](const f16* xbf, int N, int inD, const int* ei, int E,
                     const Gat& P, float* h) {
    gemm(xbf, P.Wp, P.bp, scr, N, GATH, inD);
    ln(scr, P.g0, P.b0, h, nullptr, N, GATH, 1);
    int Etot = E + N;
    k_edges<<<CDIV(Etot, 256), 256, 0, stream>>>(ei, E, N, esrc, edst);
    for (int L = 0; L < 2; ++L) {
      cvt(h, hbf, (size_t)N * GATH);
      gemm(hbf, P.Wl[L], nullptr, xl, N, GATH, GATH);
      gemm(hbf, P.Wr[L], nullptr, xr, N, GATH, GATH);
      fill(mx, -1e30f, (size_t)N * HEADS);
      fill(den, 0.f, (size_t)N * HEADS);
      k_edge_logit<<<CDIV(Etot * HEADS, 256), 256, 0, stream>>>(xl, xr, esrc, edst,
                                                               P.att[L], elog, mx, Etot);
      k_edge_exp<<<CDIV(Etot * HEADS, 256), 256, 0, stream>>>(elog, mx, edst, eex, den, Etot);
      fill(scr, 0.f, (size_t)N * GATH);
      k_edge_scatter<<<CDIV((size_t)Etot * GATH, (size_t)256), 256, 0, stream>>>(
          xl, esrc, edst, eex, den, scr, Etot);
      ln(scr, P.lg[L], P.lb[L], h, h, N, GATH, 2);
    }
  };

  // ================= pipeline =================
  ln(face_x, fin_g, fin_b, fxn, nullptr, N_FACE, 4096, 0);
  cvt(fxn, fxn_h, (size_t)N_FACE * 4096);
  cvt(context_x, ctx_h, (size_t)N_CTX * 2048);

  float* Hf = real;
  float* Hc = real + (size_t)N_FACE * GATH;
  run_gat(fxn_h, N_FACE, 4096, face_ei, E_FACE, fg, Hf);
  run_gat(ctx_h, N_CTX, 2048, ctx_ei, E_CTX, cg, Hc);

  pool(Hf, face_batch, N_FACE, GATH, poolF);
  k_head<<<CDIV(NG * NC, 256), 256, 0, stream>>>(poolF, cf_W, cf_b, out_face, GATH);
  pool(Hc, ctx_batch, N_CTX, GATH, poolC);
  k_head<<<CDIV(NG * NC, 256), 256, 0, stream>>>(poolC, cc_W, cc_b, out_ctx, GATH);

  k_rbatch<<<CDIV(NR, 256), 256, 0, stream>>>(face_batch, ctx_batch, rbatch);
  cvt(real, realbf, (size_t)NR * GATH);
  gemm(realbf, pj_W, pj_b, hlog, NR, WD, GATH);
  ln(hlog, pj_g, pj_bb, real512, nullptr, NR, WD, 1);
  cvt(real512, r512bf, (size_t)NR * WD);

  gemm(r512bf, r_W1, r_b1, hlog, NR, WD, WD);
  ln(hlog, r_g, r_bb, hlog, nullptr, NR, WD, 1);
  cvt(hlog, hlogbf, (size_t)NR * WD);
  gemm(hlogbf, r_W2, r_b2, logits4, NR, VN, WD);
  k_top2<<<CDIV(NR, 256), 256, 0, stream>>>(logits4, wts, NR);

  gemm(r512bf, m_W, m_b, msg, NR, WD, WD);
  ln(msg, m_g, m_bb, msg, nullptr, NR, WD, 1);

  k_vn_bcast<<<CDIV((size_t)NG * VN * WD, (size_t)256), 256, 0, stream>>>(
      vn_init, vn, (size_t)NG * VN * WD);
  k_vn_scatter<<<CDIV((size_t)NR * WD, (size_t)256), 256, 0, stream>>>(wts, msg, rbatch, vn, NR);

  cvt(vn, vnbf, (size_t)NG * VN * WD);
  gemm(vnbf, in_W, in_b, qkv, NG * VN, 3 * WD, WD);
  k_attn<<<CDIV(NG * HEADS * VN, 256), 256, 0, stream>>>(qkv, ovec);
  cvt(ovec, obf, (size_t)NG * VN * WD);
  gemm(obf, out_W, out_b, t512, NG * VN, WD, WD);
  k_add<<<CDIV((size_t)NG * VN * WD, (size_t)256), 256, 0, stream>>>(
      vn, t512, t512, (size_t)NG * VN * WD);
  ln(t512, n_g, n_b, vn, nullptr, NG * VN, WD, 0);

  cvt(vn, vnbf, (size_t)NG * VN * WD);
  gemm(vnbf, v_W, v_b, m2, NG * VN, WD, WD);
  ln(m2, v_g, v_bb, m2, nullptr, NG * VN, WD, 1);

  k_recv<<<CDIV((size_t)NR * WD, (size_t)256), 256, 0, stream>>>(wts, m2, rbatch, recv, NR);
  cvt(recv, recvbf, (size_t)NR * WD);
  gemm(recvbf, Wih, bih, gi, NR, 3 * WD, WD);
  gemm(r512bf, Whh, bhh, gh, NR, 3 * WD, WD);
  k_gru<<<CDIV((size_t)NR * WD, (size_t)256), 256, 0, stream>>>(gi, gh, real512, upd, NR);

  pool(upd, rbatch, NR, WD, whole);
  pool(fxn, face_batch, N_FACE, 4096, rf);
  pool(context_x, ctx_batch, N_CTX, 2048, rc);

  cvt(rf, rfbf, (size_t)NG * 4096);
  gemm(rfbf, f_W1, f_b1, rf1, NG, 1024, 4096);
  ln(rf1, f_g1, f_bb1, rf1, nullptr, NG, 1024, 1);
  cvt(rf1, rf1bf, (size_t)NG * 1024);
  gemm(rf1bf, f_W2, f_b2, rf2, NG, WD, 1024);
  ln(rf2, f_g2, f_bb2, rf2, nullptr, NG, WD, 1);

  cvt(rc, rcbf, (size_t)NG * 2048);
  gemm(rcbf, c_W, c_b, rc2, NG, WD, 2048);
  ln(rc2, c_g, c_bb, rc2, nullptr, NG, WD, 1);

  k_copy_cols<<<CDIV((size_t)NG * WD, (size_t)256), 256, 0, stream>>>(whole, comb, NG, WD, 3 * WD, 0);
  k_copy_cols<<<CDIV((size_t)NG * WD, (size_t)256), 256, 0, stream>>>(rf2, comb, NG, WD, 3 * WD, WD);
  k_copy_cols<<<CDIV((size_t)NG * WD, (size_t)256), 256, 0, stream>>>(rc2, comb, NG, WD, 3 * WD, 2 * WD);
  cvt(comb, combbf, (size_t)NG * 3 * WD);
  gemm(combbf, w_W1, w_b1, w1out, NG, WD, 3 * WD);
  ln(w1out, w_g, w_bb, w1out, nullptr, NG, WD, 1);
  k_head<<<CDIV(NG * NC, 256), 256, 0, stream>>>(w1out, w_W2, w_b2, out_whole, WD);
}
